// adaptive_style_19980187861168
// MI455X (gfx1250) — matrix operand order repaired and checked by reading, not compiled, not run on hardware
//
#include <hip/hip_runtime.h>
#include <hip/hip_bf16.h>
#include <cstdint>

// ---------------------------------------------------------------------------
// MI455X (gfx1250) adaptive-style attention, bf16 WMMA (f32 accumulate).
// GEMM B-operands staged in LDS by the Tensor Data Mover (TDM) with D# padding
// for bank-conflict-free fragment reads; fragments software-pipelined so WMMA
// overlaps VMEM instead of draining loadcnt before every matrix op.
// ---------------------------------------------------------------------------

typedef __bf16 bf16_t;
typedef unsigned int u32;
typedef __attribute__((ext_vector_type(16))) __bf16 v16bf;
typedef __attribute__((ext_vector_type(8)))  __bf16 v8bf;
typedef __attribute__((ext_vector_type(8)))  float  v8f;
typedef __attribute__((ext_vector_type(4)))  unsigned int uint4v;
typedef __attribute__((ext_vector_type(8)))  int int8v;
typedef __attribute__((ext_vector_type(4)))  int int4v;

#define NB   16
#define CCH  512
#define L1N  2048
#define L2N  1024
#define EPSF 1e-5f

// GEMM LDS B-slab: 128 rows x 256-K-half, padded pitch for bank spread.
#define KHALF  256
#define BPITCH 264   // 256 + 8 elements (TDM pad: 4 DWORDs every 128 DWORDs)

#if defined(__has_builtin)
#if __has_builtin(__builtin_amdgcn_tensor_load_to_lds) && \
    __has_builtin(__builtin_amdgcn_s_wait_tensorcnt)
#define HAVE_TDM 1
#endif
#endif
#ifndef HAVE_TDM
#define HAVE_TDM 0
#endif

// A fragment (16x32 bf16, M x K): lane m=lane&15 holds row m.
// v0..3 <- K = h*8 + 0..7 ; v4..7 <- K = 16 + h*8 + 0..7  (h = lane>>4)
__device__ __forceinline__ v16bf frag_a(const bf16_t* base, int ldk, int lane) {
  int m = lane & 15, h = lane >> 4;
  const bf16_t* p = base + (size_t)m * ldk;
  v8bf lo = *(const v8bf*)(p + h * 8);
  v8bf hi = *(const v8bf*)(p + 16 + h * 8);
  return __builtin_shufflevector(lo, hi, 0,1,2,3,4,5,6,7,8,9,10,11,12,13,14,15);
}

// B fragment (32x16 bf16, K x N) read from a K-contiguous (N x K) buffer:
// lane n=lane&15 holds column n; v0..3 <- K = h*8 + 0..7 ; v4..7 <- K = 16 + h*8 + 0..7.
__device__ __forceinline__ v16bf frag_b(const bf16_t* base, int ldk, int lane) {
  int n = lane & 15, h = lane >> 4;
  const bf16_t* p = base + (size_t)n * ldk;
  v8bf lo = *(const v8bf*)(p + h * 8);
  v8bf hi = *(const v8bf*)(p + 16 + h * 8);
  return __builtin_shufflevector(lo, hi, 0,1,2,3,4,5,6,7,8,9,10,11,12,13,14,15);
}

__device__ __forceinline__ v8f wmma_bf16(v16bf a, v16bf b, v8f c) {
  return __builtin_amdgcn_wmma_f32_16x16x32_bf16(false, a, false, b,
                                                 (short)0, c, false, false);
}

#if HAVE_TDM
// TDM 2D tile load: tileK x tileN bf16 elements, row stride strideK elements,
// LDS destination padded by 4 DWORDs every 128 DWORDs (row pitch = BPITCH).
// D# layout per CDNA5 ISA ch.8 (group0 128b, group1 256b; groups 2/3 zero).
// This toolchain exposes the 6-arg builtin:
//   (uint32x4 g0, int32x8 g1, int32x4 g2, int32x4 g3, int32x8, i32 cpol)
__device__ __forceinline__ void tdm_load_2d(u32 ldsAddr, const void* gptr,
                                            u32 tileK, u32 tileN, u32 strideK) {
  unsigned long long ga = (unsigned long long)(uintptr_t)gptr;
  uint4v g0;
  g0[0] = 1u;                                             // count=1 valid user D#
  g0[1] = ldsAddr;                                        // lds_addr (bytes)
  g0[2] = (u32)(ga & 0xffffffffu);                        // global_addr[31:0]
  g0[3] = (u32)((ga >> 32) & 0x01ffffffu) | (2u << 30);   // addr[56:32] | type=2
  int8v g1;
  // data_size=1 (2B) | pad_enable | pad_interval=6 (128 DW) | pad_amount=3 (4 DW)
  g1[0] = (int)((1u << 16) | (1u << 20) | (6u << 22) | (3u << 25));
  g1[1] = (int)((tileK & 0xffffu) << 16);                 // tensor_dim0[15:0]
  g1[2] = (int)((tileK >> 16) | ((tileN & 0xffffu) << 16)); // td0 hi | td1 lo
  g1[3] = (int)((tileN >> 16) | ((tileK & 0xffffu) << 16)); // td1 hi | tile_dim0
  g1[4] = (int)(tileN & 0xffffu);                         // tile_dim1 (tile_dim2=0)
  g1[5] = (int)strideK;                                   // tensor_dim0_stride lo
  g1[6] = 0;
  g1[7] = 0;
  int4v z4 = {0, 0, 0, 0};
  int8v z8 = {0, 0, 0, 0, 0, 0, 0, 0};
  __builtin_amdgcn_tensor_load_to_lds(g0, g1, z4, z4, z8, 0);
}
#endif

// ---------------------------------------------------------------------------
// Instance-norm statistics: one block per (n,c) row, length L. Biased var.
// ---------------------------------------------------------------------------
__global__ __launch_bounds__(256)
void stats_kernel(const float* __restrict__ src, int L,
                  float* __restrict__ mu, float* __restrict__ rs) {
  __shared__ float s1[256], s2[256];
  int row = blockIdx.x;
  const float* p = src + (size_t)row * L;
  float a = 0.f, b = 0.f;
  for (int i = threadIdx.x; i < L; i += 256) { float v = p[i]; a += v; b += v * v; }
  s1[threadIdx.x] = a; s2[threadIdx.x] = b;
  __syncthreads();
  for (int s = 128; s > 0; s >>= 1) {
    if ((int)threadIdx.x < s) {
      s1[threadIdx.x] += s1[threadIdx.x + s];
      s2[threadIdx.x] += s2[threadIdx.x + s];
    }
    __syncthreads();
  }
  if (threadIdx.x == 0) {
    float m = s1[0] / (float)L;
    float var = s2[0] / (float)L - m * m;
    mu[row] = m;
    rs[row] = rsqrtf(var + EPSF);
  }
}

// ---------------------------------------------------------------------------
// Normalize (optional) + cast to bf16 + transpose (c,L) -> (L,c) via LDS tile.
// ---------------------------------------------------------------------------
__global__ __launch_bounds__(256)
void tnorm_kernel(const float* __restrict__ src, const float* __restrict__ mu,
                  const float* __restrict__ rs, bf16_t* __restrict__ dst,
                  int L, int doNorm) {
  __shared__ bf16_t tile[32][34];
  int b = blockIdx.z;
  int l0 = blockIdx.x * 32, c0 = blockIdx.y * 32;
  int tx = threadIdx.x & 31, ty = threadIdx.x >> 5;
  for (int rr = 0; rr < 4; rr++) {
    int ccL = ty + rr * 8;
    int cc  = c0 + ccL;
    float v = src[((size_t)b * CCH + cc) * L + l0 + tx];
    if (doNorm) { int si = b * CCH + cc; v = (v - mu[si]) * rs[si]; }
    tile[ccL][tx] = (bf16_t)v;
  }
  __syncthreads();
  for (int rr = 0; rr < 4; rr++) {
    int lL = ty + rr * 8;
    dst[((size_t)b * L + l0 + lL) * CCH + c0 + tx] = tile[tx][lL];
  }
}

__global__ __launch_bounds__(256)
void castw_kernel(const float* __restrict__ src, bf16_t* __restrict__ dst, int nElem) {
  int i = blockIdx.x * 256 + threadIdx.x;
  if (i < nElem) dst[i] = (bf16_t)src[i];
}

// ---------------------------------------------------------------------------
// Batched TN GEMM: Out[m][n] = sum_k A[m][k]*B[n][k] (+bias) [+residual].
// A: (M,K) K-contig (global, register double-buffered).
// B: (N,K) K-contig, staged per 128xKHALF slab into LDS by the TDM.
// Block = 8 waves, wave tile 32x32 (2x2 WMMA), macro tile 64(M) x 128(N).
// ---------------------------------------------------------------------------
template <int FINAL>
__global__ __launch_bounds__(256)
void gemm_tn(const bf16_t* __restrict__ A, size_t strideA,
             const bf16_t* __restrict__ B, size_t strideB,
             const float* __restrict__ bias, int biasOverM,
             bf16_t* __restrict__ outB, float* __restrict__ outF,
             const float* __restrict__ resid,
             size_t strideO, int M, int N, int K) {
  __shared__ __align__(16) bf16_t Bsh[128 * BPITCH + 8];   // ~66 KB
  int lane = threadIdx.x & 31, w = threadIdx.x >> 5;
  int wm = w & 1, wn = w >> 1;
  int bb = blockIdx.z;
  int m0 = blockIdx.y * 64 + wm * 32;
  int n0blk = blockIdx.x * 128;
  const bf16_t* Ab = A + (size_t)bb * strideA + (size_t)m0 * K;
  const bf16_t* Bb = B + (size_t)bb * strideB + (size_t)n0blk * K;
  v8f acc[2][2] = {};

  for (int kh = 0; kh < K; kh += KHALF) {
    // ---- stage B slab (128 x KHALF) into LDS
#if HAVE_TDM
    if (threadIdx.x < 32) {
      tdm_load_2d((u32)(uintptr_t)&Bsh[0], Bb + kh, KHALF, 128, (u32)K);
      __builtin_amdgcn_s_wait_tensorcnt(0);
    }
#else
    for (int e = threadIdx.x; e < 128 * (KHALF / 8); e += 256) {
      int row = e >> 5, ch = e & 31;                 // 32 chunks of 8 per row
      *(v8bf*)&Bsh[row * BPITCH + ch * 8] =
          *(const v8bf*)&Bb[(size_t)row * K + kh + ch * 8];
    }
#endif
    __syncthreads();

    // ---- 8 K-steps from LDS, A fragments double-buffered from global
    v16bf a0c = frag_a(Ab + kh, K, lane);
    v16bf a1c = frag_a(Ab + 16 * K + kh, K, lane);
    for (int ks = 0; ks < 8; ks++) {
      v16bf a0n = a0c, a1n = a1c;
      if (ks < 7) {
        a0n = frag_a(Ab + kh + (ks + 1) * 32, K, lane);
        a1n = frag_a(Ab + 16 * K + kh + (ks + 1) * 32, K, lane);
      }
      v16bf b0 = frag_b(&Bsh[(wn * 32) * BPITCH + ks * 32], BPITCH, lane);
      v16bf b1 = frag_b(&Bsh[(wn * 32 + 16) * BPITCH + ks * 32], BPITCH, lane);
      acc[0][0] = wmma_bf16(a0c, b0, acc[0][0]);
      acc[0][1] = wmma_bf16(a0c, b1, acc[0][1]);
      acc[1][0] = wmma_bf16(a1c, b0, acc[1][0]);
      acc[1][1] = wmma_bf16(a1c, b1, acc[1][1]);
      a0c = a0n; a1c = a1n;
    }
    __syncthreads();   // slab consumed before next TDM overwrite
  }

  int h = lane >> 4, nn = lane & 15;
  for (int ti = 0; ti < 2; ti++)
    for (int tj = 0; tj < 2; tj++)
      for (int r = 0; r < 8; r++) {
        int gm = m0 + ti * 16 + r + h * 8;
        int gn = n0blk + wn * 32 + tj * 16 + nn;
        float v = acc[ti][tj][r] + bias[biasOverM ? gm : gn];
        size_t oi = (size_t)bb * strideO + (size_t)gm * N + gn;
        if constexpr (FINAL) outF[oi] = v + resid[oi];
        else                 outB[oi] = (bf16_t)v;
      }
}

// ---------------------------------------------------------------------------
// Attention: per (batch, 16-query tile). Scores in registers (8 v8f per wave,
// 128-wide j slice), exact softmax via shuffles + LDS reduction, probabilities
// staged in LDS, O^T GEMM. Fragments software-pipelined.
// ---------------------------------------------------------------------------
__global__ __launch_bounds__(256)
void attn_kernel(const bf16_t* __restrict__ FmT, const bf16_t* __restrict__ GmT,
                 const bf16_t* __restrict__ Hm, bf16_t* __restrict__ OT) {
  __shared__ __align__(16) bf16_t P[16 * L2N];   // probabilities, row i contiguous in j
  __shared__ float red[2][128];
  int lane = threadIdx.x & 31, w = threadIdx.x >> 5;
  int h = lane >> 4, nn = lane & 15;
  int b = blockIdx.y;
  int i0 = blockIdx.x * 16;
  const bf16_t* Fb = FmT + (size_t)b * L1N * CCH + (size_t)i0 * CCH;
  const bf16_t* Gb = GmT + (size_t)b * L2N * CCH + (size_t)(w * 128) * CCH;
  const bf16_t* Hb = Hm  + (size_t)b * CCH * L2N + (size_t)(w * 64) * L2N;

  // ---- Phase 1: S[i][j] = sum_c Fm[c][i] Gm[c][j]
  v8f acc[8] = {};
  v16bf a = frag_a(Fb, CCH, lane);
  for (int k0 = 0; k0 < CCH; k0 += 32) {
    v16bf an = a;
    if (k0 + 32 < CCH) an = frag_a(Fb + k0 + 32, CCH, lane);
    v16bf bc = frag_b(Gb + k0, CCH, lane);
    for (int t = 0; t < 8; t++) {
      v16bf bn = bc;
      if (t < 7) bn = frag_b(Gb + (size_t)((t + 1) * 16) * CCH + k0, CCH, lane);
      acc[t] = wmma_bf16(a, bc, acc[t]);
      bc = bn;
    }
    a = an;
  }

  // ---- Softmax over j (rows i = r + h*8)
  for (int r = 0; r < 8; r++) {
    float m = acc[0][r];
    for (int t = 1; t < 8; t++) m = fmaxf(m, acc[t][r]);
    for (int s = 1; s < 16; s <<= 1) m = fmaxf(m, __shfl_xor(m, s, 32));
    if (nn == 0) red[0][w * 16 + h * 8 + r] = m;
  }
  __syncthreads();
  for (int r = 0; r < 8; r++) {
    float m = red[0][h * 8 + r];
    for (int ww = 1; ww < 8; ww++) m = fmaxf(m, red[0][ww * 16 + h * 8 + r]);
    float s = 0.f;
    for (int t = 0; t < 8; t++) { float e = __expf(acc[t][r] - m); acc[t][r] = e; s += e; }
    for (int st = 1; st < 16; st <<= 1) s += __shfl_xor(s, st, 32);
    if (nn == 0) red[1][w * 16 + h * 8 + r] = s;
  }
  __syncthreads();
  for (int r = 0; r < 8; r++) {
    float s = 0.f;
    for (int ww = 0; ww < 8; ww++) s += red[1][ww * 16 + h * 8 + r];
    float inv = 1.0f / s;
    int i = h * 8 + r;
    for (int t = 0; t < 8; t++)
      P[(size_t)i * L2N + w * 128 + t * 16 + nn] = (bf16_t)(acc[t][r] * inv);
  }
  __syncthreads();

  // ---- Phase 2: O^T[i][o] = sum_j P[i][j] Hm[o][j]
  v8f oacc[4] = {};
  v16bf pa = frag_a(P, L2N, lane);
  for (int k0 = 0; k0 < L2N; k0 += 32) {
    v16bf pn = pa;
    if (k0 + 32 < L2N) pn = frag_a(P + k0 + 32, L2N, lane);   // ds_load_b128
    v16bf bc = frag_b(Hb + k0, L2N, lane);
    for (int t = 0; t < 4; t++) {
      v16bf bn = bc;
      if (t < 3) bn = frag_b(Hb + (size_t)((t + 1) * 16) * L2N + k0, L2N, lane);
      oacc[t] = wmma_bf16(pa, bc, oacc[t]);
      bc = bn;
    }
    pa = pn;
  }
  bf16_t* Ob = OT + (size_t)b * L1N * CCH + (size_t)i0 * CCH;
  for (int t = 0; t < 4; t++)
    for (int r = 0; r < 8; r++) {
      int i = r + h * 8;
      Ob[(size_t)i * CCH + w * 64 + t * 16 + nn] = (bf16_t)oacc[t][r];
    }
}

// ---------------------------------------------------------------------------
extern "C" void kernel_launch(void* const* d_in, const int* in_sizes, int n_in,
                              void* d_out, int out_size, void* d_ws, size_t ws_size,
                              hipStream_t stream) {
  const float* x   = (const float*)d_in[0];
  const float* s   = (const float*)d_in[1];
  const float* Wf  = (const float*)d_in[2];
  const float* bf_ = (const float*)d_in[3];
  const float* Wg  = (const float*)d_in[4];
  const float* bg  = (const float*)d_in[5];
  const float* Wh  = (const float*)d_in[6];
  const float* bh  = (const float*)d_in[7];
  const float* Wk  = (const float*)d_in[8];
  const float* bk  = (const float*)d_in[9];
  float* out = (float*)d_out;

  char* ws = (char*)d_ws;
  size_t off = 0;
  auto take = [&](size_t bytes) -> char* {
    char* p = ws + off;
    off += (bytes + 255) & ~(size_t)255;
    return p;
  };

  const size_t NC = (size_t)NB * CCH;
  const size_t WE = (size_t)CCH * CCH;
  float*  muX = (float*)take(NC * 4);
  float*  rsX = (float*)take(NC * 4);
  float*  muS = (float*)take(NC * 4);
  float*  rsS = (float*)take(NC * 4);
  bf16_t* Wfb = (bf16_t*)take(WE * 2);
  bf16_t* Wgb = (bf16_t*)take(WE * 2);
  bf16_t* Whb = (bf16_t*)take(WE * 2);
  bf16_t* Wkb = (bf16_t*)take(WE * 2);
  bf16_t* XnT = (bf16_t*)take((size_t)NB * L1N * CCH * 2);   // (n, L1, c)
  bf16_t* SnT = (bf16_t*)take((size_t)NB * L2N * CCH * 2);   // (n, L2, c)
  bf16_t* SrT = (bf16_t*)take((size_t)NB * L2N * CCH * 2);   // (n, L2, c)
  bf16_t* FmT = (bf16_t*)take((size_t)NB * L1N * CCH * 2);   // (n, L1, c)
  bf16_t* GmT = (bf16_t*)take((size_t)NB * L2N * CCH * 2);   // (n, L2, c)
  bf16_t* HmB = (bf16_t*)take((size_t)NB * CCH * L2N * 2);   // (n, c, L2)
  bf16_t* OT  = XnT;  // XnT dead after Fm GEMM; reuse for O^T (n, L1, c)

  stats_kernel<<<dim3((unsigned)NC), 256, 0, stream>>>(x, L1N, muX, rsX);
  stats_kernel<<<dim3((unsigned)NC), 256, 0, stream>>>(s, L2N, muS, rsS);

  castw_kernel<<<dim3((unsigned)((WE + 255) / 256)), 256, 0, stream>>>(Wf, Wfb, (int)WE);
  castw_kernel<<<dim3((unsigned)((WE + 255) / 256)), 256, 0, stream>>>(Wg, Wgb, (int)WE);
  castw_kernel<<<dim3((unsigned)((WE + 255) / 256)), 256, 0, stream>>>(Wh, Whb, (int)WE);
  castw_kernel<<<dim3((unsigned)((WE + 255) / 256)), 256, 0, stream>>>(Wk, Wkb, (int)WE);

  tnorm_kernel<<<dim3(L1N / 32, CCH / 32, NB), 256, 0, stream>>>(x, muX, rsX, XnT, L1N, 1);
  tnorm_kernel<<<dim3(L2N / 32, CCH / 32, NB), 256, 0, stream>>>(s, muS, rsS, SnT, L2N, 1);
  tnorm_kernel<<<dim3(L2N / 32, CCH / 32, NB), 256, 0, stream>>>(s, muX, rsX, SrT, L2N, 0);

  // FmT (L1 x c): M=L1, N=c, K=c; bias over N (=o)
  gemm_tn<0><<<dim3(CCH / 128, L1N / 64, NB), 256, 0, stream>>>(
      XnT, (size_t)L1N * CCH, Wfb, 0, bf_, 0, FmT, nullptr, nullptr,
      (size_t)L1N * CCH, L1N, CCH, CCH);
  // GmT (L2 x c)
  gemm_tn<0><<<dim3(CCH / 128, L2N / 64, NB), 256, 0, stream>>>(
      SnT, (size_t)L2N * CCH, Wgb, 0, bg, 0, GmT, nullptr, nullptr,
      (size_t)L2N * CCH, L2N, CCH, CCH);
  // Hm (c x L2): M=c(=o), N=L2, K=c; bias over M (=o)
  gemm_tn<0><<<dim3(L2N / 128, CCH / 64, NB), 256, 0, stream>>>(
      Whb, 0, SrT, (size_t)L2N * CCH, bh, 1, HmB, nullptr, nullptr,
      (size_t)CCH * L2N, CCH, L2N, CCH);

  // attention -> O^T (L1 x c)
  attn_kernel<<<dim3(L1N / 16, NB), 256, 0, stream>>>(FmT, GmT, HmB, OT);

  // final conv1x1 + bias + residual, f32 output (n, c, L1)
  gemm_tn<1><<<dim3(L1N / 128, CCH / 64, NB), 256, 0, stream>>>(
      Wkb, 0, OT, (size_t)L1N * CCH, bk, 1, nullptr, out, x,
      (size_t)CCH * L1N, CCH, L1N, CCH);
}
